// NeuralEmbeddingLayer_62234076119513
// MI455X (gfx1250) — matrix operand order repaired and checked by reading, not compiled, not run on hardware
//
#include <hip/hip_runtime.h>
#include <math.h>

// ---------------- problem constants ----------------
#define B_    32
#define T_    512
#define N_    1024
#define FT_   16
#define FS_   32
#define NCH_  512      // patch channels (K of GEMM1)
#define DIN_  1024     // embed dim (N of GEMM1, K of GEMM2)
#define H_    1024     // hidden (N of GEMM2)
#define NP_   1024     // patches per batch
#define MROWS 32768    // B_*NP_ rows through both GEMMs
#define SCALE_ 32.0f   // sqrt(H)

typedef __attribute__((ext_vector_type(16))) __bf16 v16bf;
typedef __attribute__((ext_vector_type(8)))  __bf16 v8bf;
typedef __attribute__((ext_vector_type(8)))  float  v8f;

__device__ __forceinline__ unsigned short f2bf(float f) {
  union { float f; unsigned int u; } v; v.f = f;
  unsigned int u = v.u;
  return (unsigned short)((u + 0x7FFFu + ((u >> 16) & 1u)) >> 16); // RNE
}

__device__ __forceinline__ float gelu_scale(float x) {
  // exact gelu * sqrt(H)
  return 0.5f * x * (1.0f + erff(x * 0.70710678118654752440f)) * SCALE_;
}

// ---- CDNA5 async global->LDS copy (ASYNCcnt path, no VGPR round trip) ----
__device__ __forceinline__ void async_g2l_b128(unsigned lds_byte_addr,
                                               const void* gptr) {
  asm volatile("global_load_async_to_lds_b128 %0, %1, off"
               :
               : "v"(lds_byte_addr),
                 "v"((unsigned long long)(uintptr_t)gptr)
               : "memory");
}
__device__ __forceinline__ void wait_async0() {
  asm volatile("s_wait_asynccnt 0x0" ::: "memory");
}

// ---------------- elementwise prep ----------------
__global__ __launch_bounds__(256) void f32_to_bf16_k(const float* __restrict__ in,
                                                     unsigned short* __restrict__ out,
                                                     int n) {
  int i = blockIdx.x * 256 + threadIdx.x;
  if (i < n) out[i] = f2bf(in[i]);
}

// spikes [B,T,N] -> patches bf16 [MROWS][NCH]; row m=b*1024+pt*32+ps, c=ft*32+fs
__global__ __launch_bounds__(256) void patchify_bf16_k(const float* __restrict__ spikes,
                                                       unsigned short* __restrict__ patches) {
  int idx = blockIdx.x * 256 + threadIdx.x;     // exactly MROWS*NCH threads
  const int m = idx >> 9;
  const int c = idx & 511;
  const int b = m >> 10;
  const int pidx = m & 1023;
  const int pt = pidx >> 5, ps = pidx & 31;
  const int ft = c >> 5,  fs = c & 31;
  const size_t src = ((size_t)(b * T_ + pt * FT_ + ft)) * N_ + (ps * FS_ + fs);
  patches[idx] = f2bf(spikes[src]);
}

__global__ void cls_gelu_k(const float* __restrict__ cls, float* __restrict__ gcls) {
  int i = blockIdx.x * 256 + threadIdx.x;
  if (i < DIN_) gcls[i] = gelu_scale(cls[i]);
}

// cls output rows: out[b,0,h] = dot(gcls, W_proj[h,:]) + b_proj[h] + pos[0,h]
__global__ __launch_bounds__(256) void cls_row_k(const float* __restrict__ gcls,
                                                 const float* __restrict__ Wp,
                                                 const float* __restrict__ bp,
                                                 const float* __restrict__ pos,
                                                 float* __restrict__ out) {
  int i = blockIdx.x * 256 + threadIdx.x;       // B_*H_ threads
  int b = i >> 10, h = i & 1023;
  float s = 0.f;
  const float* wrow = Wp + (size_t)h * DIN_;
  for (int d = 0; d < DIN_; ++d) s += gcls[d] * wrow[d];
  out[((size_t)b * 1025) * H_ + h] = s + bp[h] + pos[h];
}

__global__ void mask_stamp_k(int* __restrict__ out_i) {
  int i = blockIdx.x * 256 + threadIdx.x;
  if (i < B_ * 1025) {
    out_i[i] = 1;                    // mask
    out_i[B_ * 1025 + i] = i % 1025; // stamp
  }
}

// ---------------- WMMA GEMM core ----------------
// C[m,n] = sum_k A[m,k]*Wb[n,k]   A:[MROWS][KTOT] bf16 row-major,
// Wb:[1024][KTOT] bf16 (column n of B contiguous over k -> matches this code's
// B-matrix per-lane layout: lane holds column n=lane&15, K run per half-wave).
// A tile staged via global_load_async_to_lds_b128 (ASYNCcnt), double buffered.
// EPI==0: act = bf16(gelu(C + bias) * SCALE)    (layer-1)
// EPI==1: x[b,1+p,h] = C + bias[h] + pos[1+p,h] (layer-2, fp32 out)
template <int KTOT, int EPI>
__global__ __launch_bounds__(256) void wmma_gemm_k(const unsigned short* __restrict__ A,
                                                   const unsigned short* __restrict__ Wb,
                                                   const float* __restrict__ bias,
                                                   const float* __restrict__ pos,
                                                   unsigned short* __restrict__ actOut,
                                                   float* __restrict__ xOut) {
  __shared__ __align__(16) unsigned short sA[2][64 * 32]; // 64 rows x 32 k, double buffered

  const int bx = blockIdx.x;
  const int m0 = (bx >> 2) * 64;    // 512 M-blocks
  const int n0 = (bx & 3) * 256;    // 4 N-blocks
  const int tid  = threadIdx.x;
  const int lane = tid & 31;
  const int w    = tid >> 5;
  const int wm   = w >> 2;          // 0..1  -> 32 rows each
  const int wn   = w & 3;           // 0..3  -> 64 cols each
  const int half = lane >> 4;       // half-wave select
  const int l16  = lane & 15;

  v8f acc[2][4] = {};               // 2 M-tiles x 4 N-tiles of 16x16 f32

  // this thread's 16B slice of the A tile (256 threads cover 64x32 bf16)
  const int arow = tid >> 2;
  const int aseg = (tid & 3) * 8;   // bf16 units
  const unsigned ldsSlice =
      (unsigned)(uintptr_t)(&sA[0][0]) + (unsigned)(arow * 32 + aseg) * 2u;
  const unsigned short* gSlice = A + (size_t)(m0 + arow) * KTOT + aseg;

  // stage 0 (async memory -> LDS, no VGPR round trip)
  async_g2l_b128(ldsSlice, gSlice);
  wait_async0();
  __syncthreads();

  const int nk = KTOT / 32;
  for (int ks = 0; ks < nk; ++ks) {
    const int cur = ks & 1;
    if (ks + 1 < nk) {  // overlap: async-copy next tile into other buffer
      async_g2l_b128(ldsSlice + (unsigned)((cur ^ 1) * 64 * 32 * 2),
                     gSlice + (ks + 1) * 32);
    }
    const int k0 = ks * 32;

    // A fragments from LDS (contiguous K order: half-wave K chunk h*16..h*16+15, matches B)
    v16bf af[2];
#pragma unroll
    for (int mi = 0; mi < 2; ++mi) {
      const int rrow = wm * 32 + mi * 16 + l16;
      const v8bf lo = *(const v8bf*)(&sA[cur][rrow * 32 + half * 16]);
      const v8bf hi = *(const v8bf*)(&sA[cur][rrow * 32 + half * 16 + 8]);
      af[mi] = __builtin_shufflevector(lo, hi, 0, 1, 2, 3, 4, 5, 6, 7,
                                       8, 9, 10, 11, 12, 13, 14, 15);
    }

    // B fragments straight from global (L2-resident weights), then 8 WMMAs
#pragma unroll
    for (int ni = 0; ni < 4; ++ni) {
      const int col = n0 + wn * 64 + ni * 16 + l16;
      const unsigned short* bp = Wb + (size_t)col * KTOT + k0 + half * 16;
      if (ks + 1 < nk) __builtin_prefetch(bp + 32, 0, 1);
      const v16bf bf = *(const v16bf*)bp;
#pragma unroll
      for (int mi = 0; mi < 2; ++mi) {
        acc[mi][ni] = __builtin_amdgcn_wmma_f32_16x16x32_bf16(
            false, af[mi], false, bf, (short)0, acc[mi][ni], false, false);
      }
    }

    if (ks + 1 < nk) wait_async0();  // next tile landed in LDS
    __syncthreads();
  }

  // epilogue; C layout: VGPR r holds row (half*8 + r), col = lane&15
#pragma unroll
  for (int mi = 0; mi < 2; ++mi) {
    const int mbase = m0 + wm * 32 + mi * 16 + half * 8;
#pragma unroll
    for (int ni = 0; ni < 4; ++ni) {
      const int n = n0 + wn * 64 + ni * 16 + l16;
#pragma unroll
      for (int r = 0; r < 8; ++r) {
        const int m = mbase + r;
        float v = acc[mi][ni][r];
        if (EPI == 0) {
          v = gelu_scale(v + bias[n]);
          actOut[(size_t)m * DIN_ + n] = f2bf(v);
        } else {
          const int b = m >> 10;
          const int p = (m & 1023) + 1;   // cls occupies row 0
          v += bias[n] + pos[(size_t)p * H_ + n];
          xOut[((size_t)(b * 1025 + p)) * H_ + n] = v;
        }
      }
    }
  }
}

// ---------------- launcher ----------------
extern "C" void kernel_launch(void* const* d_in, const int* in_sizes, int n_in,
                              void* d_out, int out_size, void* d_ws, size_t ws_size,
                              hipStream_t stream) {
  (void)in_sizes; (void)n_in; (void)out_size; (void)ws_size;
  const float* spikes  = (const float*)d_in[0];
  const float* W_embed = (const float*)d_in[1];
  const float* b_embed = (const float*)d_in[2];
  const float* cls     = (const float*)d_in[3];
  const float* W_proj  = (const float*)d_in[4];
  const float* b_proj  = (const float*)d_in[5];
  const float* pos     = (const float*)d_in[6];
  float* x = (float*)d_out;

  char* ws = (char*)d_ws;
  size_t off = 0;
  auto carve = [&](size_t bytes) -> char* {
    char* p = ws + off;
    off += (bytes + 255) & ~(size_t)255;
    return p;
  };
  unsigned short* patches = (unsigned short*)carve((size_t)MROWS * NCH_ * 2); // 32 MB
  unsigned short* act     = (unsigned short*)carve((size_t)MROWS * DIN_ * 2); // 64 MB
  unsigned short* we_bf   = (unsigned short*)carve((size_t)DIN_ * NCH_ * 2);  // 1 MB
  unsigned short* wp_bf   = (unsigned short*)carve((size_t)H_ * DIN_ * 2);    // 2 MB
  float*          gcls    = (float*)carve((size_t)DIN_ * 4);

  // 1) weight / activation precision conversion + patchify
  f32_to_bf16_k<<<(DIN_ * NCH_ + 255) / 256, 256, 0, stream>>>(W_embed, we_bf, DIN_ * NCH_);
  f32_to_bf16_k<<<(H_ * DIN_ + 255) / 256, 256, 0, stream>>>(W_proj, wp_bf, H_ * DIN_);
  cls_gelu_k<<<(DIN_ + 255) / 256, 256, 0, stream>>>(cls, gcls);
  patchify_bf16_k<<<(MROWS * NCH_) / 256, 256, 0, stream>>>(spikes, patches);

  // 2) layer 1: patches x W_embed^T (+bias, gelu*scale) -> act (bf16)
  wmma_gemm_k<NCH_, 0><<<2048, 256, 0, stream>>>(patches, we_bf, b_embed, nullptr, act, nullptr);

  // 3) layer 2: act x W_proj^T (+bias, +pos) -> x rows p=1..1024
  wmma_gemm_k<DIN_, 1><<<2048, 256, 0, stream>>>(act, wp_bf, b_proj, pos, nullptr, x);

  // 4) cls rows (p=0) + integer mask/stamp outputs
  cls_row_k<<<(B_ * H_) / 256, 256, 0, stream>>>(gcls, W_proj, b_proj, pos, x);
  mask_stamp_k<<<(B_ * 1025 + 255) / 256, 256, 0, stream>>>(
      (int*)d_out + (size_t)B_ * 1025 * H_);
}
